// CausalWanTransformerBlock_90340342104724
// MI455X (gfx1250) — matrix operand order repaired and checked by reading, not compiled, not run on hardware
//
#include <hip/hip_runtime.h>
#include <hip/hip_fp16.h>

typedef _Float16 f16;
typedef __attribute__((ext_vector_type(16))) _Float16 v16h;
typedef __attribute__((ext_vector_type(8)))  float    v8f;
typedef __attribute__((ext_vector_type(4)))  unsigned int v4u;

union FragU { v4u u[2]; v16h h; };

__device__ __forceinline__ int imin(int a, int b) { return a < b ? a : b; }

// Wave-relative LDS byte offset of a __shared__ object (generic -> AS(3) cast,
// ptrtoint of an LDS pointer yields the LDS offset the async ops' VDST wants).
__device__ __forceinline__ unsigned lds_addr(const void* p) {
  return (unsigned)(unsigned long long)(const __attribute__((address_space(3))) char*)p;
}

#define EPSN 1e-6f

// ---------------------------------------------------------------------------
// Tiled WMMA GEMM:  C[M][N] = A_f16[M][K] @ Wt_f16[N][K]^T + bias, opt. GELU.
// BM=BN=128, BK=64, 256 threads = 8 waves (4 M x 2 N), wave tile 32x64.
// Tiles are staged with double-buffered GLOBAL_LOAD_ASYNC_TO_LDS_B128
// (ASYNCcnt-tracked DMA, overlapped with WMMA), both operands row-major f16
// with K contiguous so every fragment is two b128 LDS reads.
// ---------------------------------------------------------------------------
#define BM 128
#define BN 128
#define BKK 64
#define LDT 72  // BKK + 8 halves padding (row stride 144B, 16B aligned)

__global__ __launch_bounds__(256) void k_gemm(
    const f16* __restrict__ A,   // [M][K]
    const f16* __restrict__ Bt,  // [N][K]  (pre-transposed weights)
    const float* __restrict__ bias,
    float* __restrict__ Cf, f16* __restrict__ Ch,
    int M, int N, int K, int gelu)
{
  __shared__ __align__(16) f16 As[2][BM * LDT];
  __shared__ __align__(16) f16 Bs[2][BN * LDT];

  const int tid  = threadIdx.x;
  const int lane = tid & 31;
  const int wave = tid >> 5;
  const int wm   = wave & 3;   // wave row (M)
  const int wn   = wave >> 2;  // wave col (N)
  const int half = lane >> 4;
  const int l15  = lane & 15;
  const int m0   = blockIdx.y * BM;
  const int n0   = blockIdx.x * BN;

  const v8f vz = {0.f,0.f,0.f,0.f,0.f,0.f,0.f,0.f};
  v8f acc[2][4];
#pragma unroll
  for (int i = 0; i < 2; i++)
#pragma unroll
    for (int j = 0; j < 4; j++) acc[i][j] = vz;

  // Issue one BMxBKK + BNxBKK tile pair as async DMA into LDS buffer `buf`.
  auto issue_tile = [&](int k0, int buf) {
#pragma unroll
    for (int c = 0; c < 4; c++) {          // A tile: 1024 x 16B chunks
      int lin = tid * 4 + c;
      int r   = lin >> 3;
      int kk  = (lin & 7) * 8;
      int gm  = imin(m0 + r, M - 1);       // clamp: dup rows only feed dropped outputs
      const f16* ga = A + (size_t)gm * K + (k0 + kk);
      unsigned la = lds_addr(&As[buf][r * LDT + kk]);
      asm volatile("global_load_async_to_lds_b128 %0, %1, off"
                   :: "v"(la), "v"(ga) : "memory");
    }
#pragma unroll
    for (int c = 0; c < 4; c++) {          // B tile: 1024 x 16B chunks
      int lin = tid * 4 + c;
      int r   = lin >> 3;
      int kk  = (lin & 7) * 8;
      const f16* gb = Bt + (size_t)(n0 + r) * K + (k0 + kk);
      unsigned lb = lds_addr(&Bs[buf][r * LDT + kk]);
      asm volatile("global_load_async_to_lds_b128 %0, %1, off"
                   :: "v"(lb), "v"(gb) : "memory");
    }
  };

  issue_tile(0, 0);
  asm volatile("s_wait_asynccnt 0x0" ::: "memory");
  __syncthreads();

  int buf = 0;
  for (int k0 = 0; k0 < K; k0 += BKK) {
    if (k0 + BKK < K) issue_tile(k0 + BKK, buf ^ 1);  // overlap DMA with WMMA

#pragma unroll
    for (int ks = 0; ks < BKK / 32; ks++) {
      const int kb = ks * 32;
      FragU a[2];
#pragma unroll
      for (int mi = 0; mi < 2; mi++) {
        int row = wm * 32 + mi * 16 + l15;
        const f16* p = &As[buf][row * LDT + kb + 8 * half];
        a[mi].u[0] = *(const v4u*)p;
        a[mi].u[1] = *(const v4u*)(p + 16);
      }
#pragma unroll
      for (int ni = 0; ni < 4; ni++) {
        int col = wn * 64 + ni * 16 + l15;
        const f16* p = &Bs[buf][col * LDT + kb + 8 * half];
        FragU b;
        b.u[0] = *(const v4u*)p;
        b.u[1] = *(const v4u*)(p + 16);
#pragma unroll
        for (int mi = 0; mi < 2; mi++)
          acc[mi][ni] = __builtin_amdgcn_wmma_f32_16x16x32_f16(
              false, a[mi].h, false, b.h, (short)0, acc[mi][ni], false, false);
      }
    }
    asm volatile("s_wait_asynccnt 0x0" ::: "memory");  // own DMAs landed
    __syncthreads();                                   // everyone's DMAs landed
    buf ^= 1;
  }

  // ---- epilogue: bias (+GELU-tanh), store f32 or f16 ----
#pragma unroll
  for (int ni = 0; ni < 4; ni++) {
    int col = n0 + wn * 64 + ni * 16 + l15;
    float bv = bias ? bias[col] : 0.0f;
#pragma unroll
    for (int mi = 0; mi < 2; mi++) {
#pragma unroll
      for (int r = 0; r < 8; r++) {
        int row = m0 + wm * 32 + mi * 16 + r + 8 * half;
        if (row < M) {
          float v = acc[mi][ni][r] + bv;
          if (gelu) {
            float x3 = v * v * v;
            v = 0.5f * v * (1.0f + tanhf(0.7978845608028654f * (v + 0.044715f * x3)));
          }
          if (Ch) Ch[(size_t)row * N + col] = (f16)v;
          else    Cf[(size_t)row * N + col] = v;
        }
      }
    }
  }
}

// ---------------------------------------------------------------------------
// Weight prep: W f32 [K][N]  ->  Wt f16 [N][K]   (32x32 tiles via LDS)
// Done once per launch; f16 weight set (~93MB) fits MI455X's 192MB L2, so
// the 25 M-tile GEMM passes re-read weights from L2, not HBM.
// ---------------------------------------------------------------------------
__global__ __launch_bounds__(256) void k_cvtT(
    const float* __restrict__ W, f16* __restrict__ Wt, int K, int N)
{
  __shared__ float tile[32][33];
  const int tx = threadIdx.x & 31;
  const int ty = threadIdx.x >> 5;      // 0..7
  const int nb = blockIdx.x * 32;       // N offset
  const int kb = blockIdx.y * 32;       // K offset
#pragma unroll
  for (int j = 0; j < 4; j++)
    tile[ty + 8 * j][tx] = W[(size_t)(kb + ty + 8 * j) * N + (nb + tx)];
  __syncthreads();
#pragma unroll
  for (int j = 0; j < 4; j++)
    Wt[(size_t)(nb + ty + 8 * j) * K + (kb + tx)] = (f16)tile[tx][ty + 8 * j];
}

// ---------------------------------------------------------------------------
// Flash attention (hd = 128). Q: [H][Lq][128] f16, K: [H][Lk][128] f16,
// V transposed: [H][128][Lk] f16. Output (f16) written as [Lq][H*128].
// 8 independent waves per block; each wave owns one 16-query tile and walks
// keys in 32-wide steps with online softmax. Frame-causal mask from FS.
// ---------------------------------------------------------------------------
__global__ __launch_bounds__(256) void k_attn(
    const f16* __restrict__ Qh, const f16* __restrict__ Kh,
    const f16* __restrict__ Vt, f16* __restrict__ O,
    int Lq, int Lk, int H, int FS, int causal)
{
  __shared__ __align__(16) f16 Pbuf[8][16 * 32];
  const int tid  = threadIdx.x;
  const int lane = tid & 31;
  const int wave = tid >> 5;
  const int half = lane >> 4;
  const int l15  = lane & 15;
  const int h    = blockIdx.y;
  const int qbase = blockIdx.x * 128 + wave * 16;
  if (qbase >= Lq) return;  // wave-uniform exit; no cross-wave barriers used

  const float scale = 0.08838834764831845f;  // 1/sqrt(128)
  const v8f vz = {0.f,0.f,0.f,0.f,0.f,0.f,0.f,0.f};
  const v4u uz = {0u,0u,0u,0u};

  int qrow = qbase + l15;
  if (qrow >= Lq) qrow = Lq - 1;  // clamp; contributions masked, stores guarded
  const f16* qp = Qh + ((size_t)h * Lq + qrow) * 128;
  FragU qf[4];
#pragma unroll
  for (int kc = 0; kc < 4; kc++) {
    int off = kc * 32 + 8 * half;
    qf[kc].u[0] = *(const v4u*)(qp + off);
    qf[kc].u[1] = *(const v4u*)(qp + off + 16);
  }

  float m_i[8], l_i[8], alpha[8];
  v8f o[8];
#pragma unroll
  for (int r = 0; r < 8; r++) { m_i[r] = -1e30f; l_i[r] = 0.0f; }
#pragma unroll
  for (int n = 0; n < 8; n++) o[n] = vz;

  const int qlast = imin(qbase + 15, Lq - 1);
  const int kmax  = causal ? imin(Lk, ((qlast / FS) + 1) * FS) : Lk;

  for (int kt = 0; kt < kmax; kt += 32) {
    // ---- scores: S = Q @ K^T (two 16-key subtiles) ----
    v8f s[2] = {vz, vz};
#pragma unroll
    for (int st = 0; st < 2; st++) {
      int key = kt + st * 16 + l15;
      bool kv = key < Lk;
      const f16* kp = Kh + ((size_t)h * Lk + (kv ? key : 0)) * 128;
#pragma unroll
      for (int kc = 0; kc < 4; kc++) {
        FragU bf;
        int off = kc * 32 + 8 * half;
        bf.u[0] = kv ? *(const v4u*)(kp + off)     : uz;
        bf.u[1] = kv ? *(const v4u*)(kp + off + 16) : uz;
        s[st] = __builtin_amdgcn_wmma_f32_16x16x32_f16(
            false, qf[kc].h, false, bf.h, (short)0, s[st], false, false);
      }
    }
    // ---- scale + mask + online softmax stats; stage P into LDS (f16) ----
#pragma unroll
    for (int r = 0; r < 8; r++) {
      int qr  = qbase + r + 8 * half;
      bool qv = qr < Lq;
      int qfr = (qv ? qr : 0) / FS;
      int key0 = kt + l15;
      int key1 = kt + 16 + l15;
      bool ok0 = qv && key0 < Lk && (!causal || key0 / FS <= qfr);
      bool ok1 = qv && key1 < Lk && (!causal || key1 / FS <= qfr);
      float v0 = ok0 ? s[0][r] * scale : -1e30f;
      float v1 = ok1 ? s[1][r] * scale : -1e30f;
      float rowm = fmaxf(v0, v1);
#pragma unroll
      for (int ofs = 1; ofs < 16; ofs <<= 1)
        rowm = fmaxf(rowm, __shfl_xor(rowm, ofs, 16));
      float mn = fmaxf(m_i[r], rowm);
      float p0 = __expf(v0 - mn);
      float p1 = __expf(v1 - mn);
      float rs = p0 + p1;
#pragma unroll
      for (int ofs = 1; ofs < 16; ofs <<= 1)
        rs += __shfl_xor(rs, ofs, 16);
      alpha[r] = __expf(m_i[r] - mn);
      l_i[r]   = l_i[r] * alpha[r] + rs;
      m_i[r]   = mn;
      int prow = r + 8 * half;
      Pbuf[wave][prow * 32 + l15]      = (f16)p0;
      Pbuf[wave][prow * 32 + 16 + l15] = (f16)p1;
    }
    // rescale running output accumulators
#pragma unroll
    for (int n = 0; n < 8; n++)
#pragma unroll
      for (int r = 0; r < 8; r++) o[n][r] = o[n][r] * alpha[r];

    // wave-level DS RAW fence before re-reading P in A-fragment layout
    asm volatile("s_wait_dscnt 0x0" ::: "memory");

    FragU pf;
    {
      const f16* pp = &Pbuf[wave][l15 * 32 + 8 * half];
      pf.u[0] = *(const v4u*)pp;
      pf.u[1] = *(const v4u*)(pp + 16);
    }
    // ---- O += P @ V ----
#pragma unroll
    for (int n = 0; n < 8; n++) {
      int hdcol = n * 16 + l15;
      const f16* vp = Vt + ((size_t)h * 128 + hdcol) * Lk + kt + 8 * half;
      bool vv = (kt + 8 * half) < Lk;
      FragU vf;
      vf.u[0] = vv ? *(const v4u*)vp       : uz;
      vf.u[1] = vv ? *(const v4u*)(vp + 16) : uz;
      o[n] = __builtin_amdgcn_wmma_f32_16x16x32_f16(
          false, pf.h, false, vf.h, (short)0, o[n], false, false);
    }
  }

  // ---- normalize and store [Lq][H*128] f16 ----
#pragma unroll
  for (int n = 0; n < 8; n++) {
    int col = h * 128 + n * 16 + l15;
#pragma unroll
    for (int r = 0; r < 8; r++) {
      int row = qbase + r + 8 * half;
      if (row < Lq) {
        float inv = l_i[r] > 0.f ? 1.0f / l_i[r] : 0.0f;
        O[(size_t)row * (H * 128) + col] = (f16)(o[n][r] * inv);
      }
    }
  }
}

// ---------------------------------------------------------------------------
// Elementwise / normalization kernels
// ---------------------------------------------------------------------------
__global__ __launch_bounds__(256) void k_ln_mod(
    const float* __restrict__ x, const float* __restrict__ sst,
    const float* __restrict__ temb, int sidx, int cidx,
    f16* __restrict__ out, int D, int FS)
{
  __shared__ float red[256];
  const int row = blockIdx.x;
  const int f = row / FS;
  const float* xr = x + (size_t)row * D;
  float s = 0.f, sq = 0.f;
  for (int d = threadIdx.x; d < D; d += 256) { float v = xr[d]; s += v; sq += v * v; }
  red[threadIdx.x] = s; __syncthreads();
  for (int o = 128; o > 0; o >>= 1) { if (threadIdx.x < o) red[threadIdx.x] += red[threadIdx.x + o]; __syncthreads(); }
  float mean = red[0] / D; __syncthreads();
  red[threadIdx.x] = sq; __syncthreads();
  for (int o = 128; o > 0; o >>= 1) { if (threadIdx.x < o) red[threadIdx.x] += red[threadIdx.x + o]; __syncthreads(); }
  float var = red[0] / D - mean * mean;
  float inv = rsqrtf(var + EPSN);
  const float* sb  = sst + (size_t)sidx * D;
  const float* cb  = sst + (size_t)cidx * D;
  const float* tbs = temb + ((size_t)f * 6 + sidx) * D;
  const float* tbc = temb + ((size_t)f * 6 + cidx) * D;
  for (int d = threadIdx.x; d < D; d += 256) {
    float sc = cb[d] + tbc[d];
    float sh = sb[d] + tbs[d];
    out[(size_t)row * D + d] = (f16)(((xr[d] - mean) * inv) * (1.0f + sc) + sh);
  }
}

__global__ __launch_bounds__(256) void k_ln_affine(
    const float* __restrict__ x, const float* __restrict__ g,
    const float* __restrict__ b, f16* __restrict__ out, int D)
{
  __shared__ float red[256];
  const int row = blockIdx.x;
  const float* xr = x + (size_t)row * D;
  float s = 0.f, sq = 0.f;
  for (int d = threadIdx.x; d < D; d += 256) { float v = xr[d]; s += v; sq += v * v; }
  red[threadIdx.x] = s; __syncthreads();
  for (int o = 128; o > 0; o >>= 1) { if (threadIdx.x < o) red[threadIdx.x] += red[threadIdx.x + o]; __syncthreads(); }
  float mean = red[0] / D; __syncthreads();
  red[threadIdx.x] = sq; __syncthreads();
  for (int o = 128; o > 0; o >>= 1) { if (threadIdx.x < o) red[threadIdx.x] += red[threadIdx.x + o]; __syncthreads(); }
  float var = red[0] / D - mean * mean;
  float inv = rsqrtf(var + EPSN);
  for (int d = threadIdx.x; d < D; d += 256)
    out[(size_t)row * D + d] = (f16)((xr[d] - mean) * inv * g[d] + b[d]);
}

// RMSNorm over D (+ optional interleaved RoPE), write f16 [H][Lrows][hd]
__global__ __launch_bounds__(256) void k_rms_rope(
    const float* __restrict__ x, const float* __restrict__ w,
    const float* __restrict__ cosb, const float* __restrict__ sinb,
    f16* __restrict__ out, int Lrows, int D, int hd, int do_rope)
{
  __shared__ float red[256];
  const int row = blockIdx.x;
  const float* xr = x + (size_t)row * D;
  float sq = 0.f;
  for (int d = threadIdx.x; d < D; d += 256) { float v = xr[d]; sq += v * v; }
  red[threadIdx.x] = sq; __syncthreads();
  for (int o = 128; o > 0; o >>= 1) { if (threadIdx.x < o) red[threadIdx.x] += red[threadIdx.x + o]; __syncthreads(); }
  float inv = rsqrtf(red[0] / D + EPSN);
  const int P = D / 2, hp = hd / 2;
  for (int p = threadIdx.x; p < P; p += 256) {
    int i0 = 2 * p, i1 = i0 + 1;
    float y0 = xr[i0] * inv * w[i0];
    float y1 = xr[i1] * inv * w[i1];
    int head = i0 / hd, dh = i0 % hd;
    float o0 = y0, o1 = y1;
    if (do_rope) {
      float c = cosb[(size_t)row * hp + (dh >> 1)];
      float s = sinb[(size_t)row * hp + (dh >> 1)];
      o0 = y0 * c - y1 * s;
      o1 = y0 * s + y1 * c;
    }
    size_t base = ((size_t)head * Lrows + row) * hd + dh;
    out[base]     = (f16)o0;
    out[base + 1] = (f16)o1;
  }
}

// V: f32 [L][D] -> f16 [H][hd][L]  (hd-major for WMMA B fragments)
__global__ void k_prep_v(const float* __restrict__ x, f16* __restrict__ out,
                         int Lrows, int D, int hd)
{
  size_t i = (size_t)blockIdx.x * blockDim.x + threadIdx.x;
  size_t n = (size_t)Lrows * D;
  if (i >= n) return;
  int row = (int)(i / D), c = (int)(i % D);
  int head = c / hd, dh = c % hd;
  out[((size_t)head * hd + dh) * Lrows + row] = (f16)x[i];
}

// out = base + delta * gate ; gate = sst[g]+temb[f][g] per frame, or 1 if g<0
__global__ void k_gate_add(const float* __restrict__ base,
                           const float* __restrict__ delta,
                           const float* __restrict__ sst,
                           const float* __restrict__ temb, int gidx,
                           float* __restrict__ out, int D, int FS, size_t n)
{
  size_t i = (size_t)blockIdx.x * blockDim.x + threadIdx.x;
  if (i >= n) return;
  int row = (int)(i / D), d = (int)(i % D);
  float g = 1.0f;
  if (gidx >= 0) {
    int f = row / FS;
    g = sst[(size_t)gidx * D + d] + temb[((size_t)f * 6 + gidx) * D + d];
  }
  out[i] = base[i] + delta[i] * g;
}

__global__ void k_cvt_f32_f16(const float* __restrict__ x, f16* __restrict__ y, size_t n)
{
  size_t i = (size_t)blockIdx.x * blockDim.x + threadIdx.x;
  if (i < n) y[i] = (f16)x[i];
}

// ---------------------------------------------------------------------------
// Host orchestration
// ---------------------------------------------------------------------------
extern "C" void kernel_launch(void* const* d_in, const int* in_sizes, int n_in,
                              void* d_out, int out_size, void* d_ws, size_t ws_size,
                              hipStream_t stream)
{
  const float* hidden = (const float*)d_in[0];
  const float* enc    = (const float*)d_in[1];
  const float* temb   = (const float*)d_in[2];
  const float* cosb   = (const float*)d_in[3];
  const float* sinb   = (const float*)d_in[4];
  // d_in[5] = attn_mask (bool) -- mask recomputed analytically from FS
  const float* sst    = (const float*)d_in[6];
  const float* wq = (const float*)d_in[7];
  const float* wk = (const float*)d_in[8];
  const float* wv = (const float*)d_in[9];
  const float* wo = (const float*)d_in[10];
  const float* bq = (const float*)d_in[11];
  const float* bk = (const float*)d_in[12];
  const float* bv = (const float*)d_in[13];
  const float* bo = (const float*)d_in[14];
  const float* nq_w  = (const float*)d_in[15];
  const float* nk_w  = (const float*)d_in[16];
  const float* ln2_g = (const float*)d_in[17];
  const float* ln2_b = (const float*)d_in[18];
  const float* cwq = (const float*)d_in[19];
  const float* cwk = (const float*)d_in[20];
  const float* cwv = (const float*)d_in[21];
  const float* cwo = (const float*)d_in[22];
  const float* cbq = (const float*)d_in[23];
  const float* cbk = (const float*)d_in[24];
  const float* cbv = (const float*)d_in[25];
  const float* cbo = (const float*)d_in[26];
  const float* cnq_w = (const float*)d_in[27];
  const float* cnk_w = (const float*)d_in[28];
  const float* w1 = (const float*)d_in[29];
  const float* b1 = (const float*)d_in[30];
  const float* w2 = (const float*)d_in[31];
  const float* b2 = (const float*)d_in[32];

  const int D  = in_sizes[6] / 6;
  const int L  = in_sizes[0] / D;
  const int LC = in_sizes[1] / D;
  const int F  = in_sizes[2] / (6 * D);
  const int FS = L / F;
  const int FF = in_sizes[29] / D;
  const int hd = (2 * in_sizes[3]) / L;   // 128
  const int H  = D / hd;

  char* ws = (char*)d_ws;
  size_t off = 0;
  auto wsa = [&](size_t bytes) -> char* {
    char* p = ws + off;
    off = (off + bytes + 255) & ~(size_t)255;
    return p;
  };
  // pre-transposed f16 weights
  f16* wqT  = (f16*)wsa((size_t)D * D * sizeof(f16));
  f16* wkT  = (f16*)wsa((size_t)D * D * sizeof(f16));
  f16* wvT  = (f16*)wsa((size_t)D * D * sizeof(f16));
  f16* woT  = (f16*)wsa((size_t)D * D * sizeof(f16));
  f16* cwqT = (f16*)wsa((size_t)D * D * sizeof(f16));
  f16* cwkT = (f16*)wsa((size_t)D * D * sizeof(f16));
  f16* cwvT = (f16*)wsa((size_t)D * D * sizeof(f16));
  f16* cwoT = (f16*)wsa((size_t)D * D * sizeof(f16));
  f16* w1T  = (f16*)wsa((size_t)D * FF * sizeof(f16)); // [FF][D]
  f16* w2T  = (f16*)wsa((size_t)FF * D * sizeof(f16)); // [D][FF]
  // activations
  f16*   nh   = (f16*)  wsa((size_t)L * D * sizeof(f16));
  f16*   ench = (f16*)  wsa((size_t)LC * D * sizeof(f16));
  float* tmp  = (float*)wsa((size_t)L * D * sizeof(float));
  f16*   qh   = (f16*)  wsa((size_t)L * D * sizeof(f16));
  f16*   kh   = (f16*)  wsa((size_t)L * D * sizeof(f16));
  f16*   vt   = (f16*)  wsa((size_t)L * D * sizeof(f16));
  f16*   attn = (f16*)  wsa((size_t)L * D * sizeof(f16));
  float* hbuf = (float*)wsa((size_t)L * D * sizeof(float));
  f16*   ff1  = (f16*)  wsa((size_t)L * FF * sizeof(f16));
  (void)ws_size; (void)n_in; (void)out_size;

  const dim3 blk(256);
  const size_t nLD = (size_t)L * D;
  auto gs = [&](size_t n) { return dim3((unsigned)((n + 255) / 256)); };
  auto cvtT = [&](const float* W, f16* Wt, int K, int N) {
    dim3 grid((unsigned)(N / 32), (unsigned)(K / 32));
    k_cvtT<<<grid, blk, 0, stream>>>(W, Wt, K, N);
  };
  auto gemm = [&](const f16* A, const f16* Wt, const float* bias,
                  float* Cf, f16* Ch, int M, int N, int K, int gelu) {
    dim3 grid((unsigned)(N / 128), (unsigned)((M + 127) / 128));
    k_gemm<<<grid, blk, 0, stream>>>(A, Wt, bias, Cf, Ch, M, N, K, gelu);
  };

  // weight prep (f32 [K][N] -> f16 [N][K]); ~93MB total, fits in 192MB L2
  cvtT(wq,  wqT,  D, D);  cvtT(wk,  wkT,  D, D);
  cvtT(wv,  wvT,  D, D);  cvtT(wo,  woT,  D, D);
  cvtT(cwq, cwqT, D, D);  cvtT(cwk, cwkT, D, D);
  cvtT(cwv, cwvT, D, D);  cvtT(cwo, cwoT, D, D);
  cvtT(w1,  w1T,  D, FF); cvtT(w2,  w2T,  FF, D);

  // encoder hidden -> f16
  k_cvt_f32_f16<<<gs((size_t)LC * D), blk, 0, stream>>>(enc, ench, (size_t)LC * D);

  // nh = LN(hidden)*(1+scale_msa)+shift_msa   (indices 0,1)
  k_ln_mod<<<dim3(L), blk, 0, stream>>>(hidden, sst, temb, 0, 1, nh, D, FS);

  // Q / K / V projections + norms + rope
  gemm(nh, wqT, bq, tmp, nullptr, L, D, D, 0);
  k_rms_rope<<<dim3(L), blk, 0, stream>>>(tmp, nq_w, cosb, sinb, qh, L, D, hd, 1);
  gemm(nh, wkT, bk, tmp, nullptr, L, D, D, 0);
  k_rms_rope<<<dim3(L), blk, 0, stream>>>(tmp, nk_w, cosb, sinb, kh, L, D, hd, 1);
  gemm(nh, wvT, bv, tmp, nullptr, L, D, D, 0);
  k_prep_v<<<gs(nLD), blk, 0, stream>>>(tmp, vt, L, D, hd);

  // self attention (frame-causal)
  {
    dim3 grid((unsigned)((L + 127) / 128), (unsigned)H);
    k_attn<<<grid, blk, 0, stream>>>(qh, kh, vt, attn, L, L, H, FS, 1);
  }
  gemm(attn, woT, bo, tmp, nullptr, L, D, D, 0);
  // h = hidden + attn_out * gate_msa (index 2)
  k_gate_add<<<gs(nLD), blk, 0, stream>>>(hidden, tmp, sst, temb, 2, hbuf, D, FS, nLD);

  // norm_h = LN(h)*ln2_g + ln2_b
  k_ln_affine<<<dim3(L), blk, 0, stream>>>(hbuf, ln2_g, ln2_b, nh, D);

  // cross attention
  gemm(nh, cwqT, cbq, tmp, nullptr, L, D, D, 0);
  k_rms_rope<<<dim3(L), blk, 0, stream>>>(tmp, cnq_w, cosb, sinb, qh, L, D, hd, 0);
  gemm(ench, cwkT, cbk, tmp, nullptr, LC, D, D, 0);
  k_rms_rope<<<dim3(LC), blk, 0, stream>>>(tmp, cnk_w, cosb, sinb, kh, LC, D, hd, 0);
  gemm(ench, cwvT, cbv, tmp, nullptr, LC, D, D, 0);
  k_prep_v<<<gs((size_t)LC * D), blk, 0, stream>>>(tmp, vt, LC, D, hd);
  {
    dim3 grid((unsigned)((L + 127) / 128), (unsigned)H);
    k_attn<<<grid, blk, 0, stream>>>(qh, kh, vt, attn, L, LC, H, FS, 0);
  }
  gemm(attn, cwoT, cbo, tmp, nullptr, L, D, D, 0);
  // h = h + cross_out
  k_gate_add<<<gs(nLD), blk, 0, stream>>>(hbuf, tmp, nullptr, nullptr, -1, hbuf, D, FS, nLD);

  // nh = LN(h)*(1+c_scale)+c_shift  (indices 3,4)
  k_ln_mod<<<dim3(L), blk, 0, stream>>>(hbuf, sst, temb, 3, 4, nh, D, FS);

  // FFN: gelu(nh @ w1 + b1) @ w2 + b2
  gemm(nh, w1T, b1, nullptr, ff1, L, FF, D, 1);
  gemm(ff1, w2T, b2, tmp, nullptr, L, D, FF, 0);

  // out = h + ff * c_gate (index 5)
  k_gate_add<<<gs(nLD), blk, 0, stream>>>(hbuf, tmp, sst, temb, 5, (float*)d_out, D, FS, nLD);
}
